// GARCH_GRU_16690242913271
// MI455X (gfx1250) — matrix operand order repaired and checked by reading, not compiled, not run on hardware
//
#include <hip/hip_runtime.h>

typedef __attribute__((ext_vector_type(16))) _Float16 v16h;
typedef __attribute__((ext_vector_type(8)))  _Float16 v8h;
typedef __attribute__((ext_vector_type(8)))  float    v8f;
typedef __attribute__((ext_vector_type(4)))  float    v4f;

#define TSTEPS 1024
#define NW     4            // waves per block
#define RPW    16           // batch rows per wave (WMMA M)
#define HP     72           // padded row pitch (halves) for sW / sH  -> bank-conflict-free
#define XP     36           // padded row pitch (floats) for staged x
#define NCOL   192          // padded gate columns: 3 gates x 64

__device__ __forceinline__ float fast_sigmoid(float x) {
    float e = __builtin_amdgcn_exp2f(-1.442695041f * x);
    return __builtin_amdgcn_rcpf(1.0f + e);
}
__device__ __forceinline__ float fast_tanh(float x) {
    float e = __builtin_amdgcn_exp2f(-2.885390082f * x);
    return 2.0f * __builtin_amdgcn_rcpf(1.0f + e) - 1.0f;
}

union AFrag { v16h v; struct { v8h lo, hi; } p; };

__global__ __launch_bounds__(NW * 32)
void garch_gru_kernel(const float* __restrict__ x,        // (B, T) row-major
                      const float* __restrict__ omega_p,
                      const float* __restrict__ alpha_p,
                      const float* __restrict__ beta_p,
                      const float* __restrict__ W_ih,     // (150, 2)
                      const float* __restrict__ W_hh,     // (150, 50)
                      const float* __restrict__ b_ih,     // (150)
                      const float* __restrict__ b_hh,     // (150)
                      const float* __restrict__ fc_w,     // (50)
                      const float* __restrict__ fc_b,     // (1)
                      float* __restrict__ out)            // (B)
{
    // W_hh^T in f16, B-operand friendly: sW[c*HP + k] = W_hh[row(c)][k], zero padded
    __shared__ _Float16 sW[NCOL * HP];
    __shared__ float sWih0[NCOL], sWih1[NCOL], sBih[NCOL], sBsum[NCOL], sBhh[NCOL];
    __shared__ float sFc[64];
    __shared__ _Float16 sH[NW][RPW * HP];    // per-wave hidden state (f16, padded)
    __shared__ float    sX[NW][RPW * XP];    // per-wave staged x chunk (32 steps)
    __shared__ float    sS2[NW][RPW];        // per-wave sigma^2 state

    const int tid = threadIdx.x;

    // ---- init shared state ----
    for (int idx = tid; idx < NCOL * 64; idx += NW * 32) {
        int c = idx >> 6, k = idx & 63;
        int g = c >> 6, j = c & 63;          // gate g (0=r,1=z,2=n), in-gate col j
        float v = 0.f;
        if (j < 50 && k < 50) v = W_hh[(g * 50 + j) * 50 + k];
        sW[c * HP + k] = (_Float16)v;
    }
    for (int c = tid; c < NCOL; c += NW * 32) {
        int g = c >> 6, j = c & 63;
        float w0 = 0.f, w1 = 0.f, bi = 0.f, bh = 0.f;
        if (j < 50) { int r = g * 50 + j; w0 = W_ih[r*2]; w1 = W_ih[r*2+1]; bi = b_ih[r]; bh = b_hh[r]; }
        sWih0[c] = w0; sWih1[c] = w1; sBih[c] = bi; sBhh[c] = bh; sBsum[c] = bi + bh;
    }
    for (int k = tid; k < 64; k += NW * 32) sFc[k] = (k < 50) ? fc_w[k] : 0.f;
    for (int idx = tid; idx < NW * RPW * HP; idx += NW * 32) (&sH[0][0])[idx] = (_Float16)0.f;
    for (int idx = tid; idx < NW * RPW; idx += NW * 32) (&sS2[0][0])[idx] = 0.f;
    __syncthreads();

    const int wave = tid >> 5;
    const int lane = tid & 31;
    const int lm   = lane & 15;      // column / row-in-tile index
    const int kh   = lane >> 4;      // K-half select per WMMA lane layout
    const int rowbase = blockIdx.x * (NW * RPW) + wave * RPW;

    const float omega = omega_p[0], alpha = alpha_p[0], beta = beta_p[0];

    _Float16* hbuf  = sH[wave];
    float*    xbuf  = sX[wave];
    float*    s2buf = sS2[wave];

    float rF[4][8], zF[4][8];
    float hD[4][8];                  // h in C/D fragment layout: M = v + 8*kh, col = j*16+lm
    #pragma unroll
    for (int j = 0; j < 4; ++j)
        #pragma unroll
        for (int v = 0; v < 8; ++v) hD[j][v] = 0.f;

    for (int t = 0; t < TSTEPS; ++t) {
        if ((t & 31) == 0) {
            // stage next 32 timesteps of x for this wave's 16 rows
            const float* gp = x + (size_t)(rowbase + lm) * TSTEPS + t + kh * 16;
            float* lp = xbuf + lm * XP + kh * 16;
            #pragma unroll
            for (int i = 0; i < 4; ++i) *(v4f*)(lp + 4*i) = *(const v4f*)(gp + 4*i);
        }
        const int tt = t & 31;

        // sigma^2 recursion (per batch row, lanes 0..15)
        if (lane < 16) {
            float xv = xbuf[lane * XP + tt];
            s2buf[lane] = omega + alpha * xv * xv + beta * s2buf[lane];
        }
        // per-lane copies of x, sigma^2 for the 8 C-fragment rows (M = kh*8 + v)
        float xr[8], s2r[8];
        #pragma unroll
        for (int v = 0; v < 8; ++v) {
            int r = kh * 8 + v;
            xr[v]  = xbuf[r * XP + tt];
            s2r[v] = s2buf[r];
        }
        // A fragments (h, f16): 16x32 layout — lane holds row lm, K-halves per kh
        AFrag a0, a1;
        {
            const _Float16* hp = hbuf + lm * HP;
            a0.p.lo = *(const v8h*)(hp + kh * 16);       // K = 16*kh + 0..7
            a0.p.hi = *(const v8h*)(hp + kh * 16 + 8);   // K = 16*kh + 8..15
            a1.p.lo = *(const v8h*)(hp + 32 + kh * 16);
            a1.p.hi = *(const v8h*)(hp + 32 + kh * 16 + 8);
        }

        // ---- r and z gate tiles (cols 0..127 of padded space) ----
        #pragma unroll
        for (int tl = 0; tl < 8; ++tl) {
            int c = tl * 16 + lm;
            AFrag b0, b1;
            const _Float16* wp = sW + c * HP;
            b0.p.lo = *(const v8h*)(wp + kh * 16);       // K = 16*kh + 0..7
            b0.p.hi = *(const v8h*)(wp + kh * 16 + 8);   // K = 16*kh + 8..15
            b1.p.lo = *(const v8h*)(wp + 32 + kh * 16);
            b1.p.hi = *(const v8h*)(wp + 32 + kh * 16 + 8);
            v8f acc;
            float w0 = sWih0[c], w1 = sWih1[c], bs = sBsum[c];
            #pragma unroll
            for (int v = 0; v < 8; ++v)
                acc[v] = fmaf(xr[v], w0, fmaf(s2r[v], w1, bs));   // gi + b_ih + b_hh
            acc = __builtin_amdgcn_wmma_f32_16x16x32_f16(false, a0.v, false, b0.v, (short)0, acc, false, false);
            acc = __builtin_amdgcn_wmma_f32_16x16x32_f16(false, a1.v, false, b1.v, (short)0, acc, false, false);
            if (tl < 4) {
                #pragma unroll
                for (int v = 0; v < 8; ++v) rF[tl][v] = fast_sigmoid(acc[v]);
            } else {
                #pragma unroll
                for (int v = 0; v < 8; ++v) zF[tl - 4][v] = fast_sigmoid(acc[v]);
            }
        }

        // ---- n gate tiles + hidden-state update ----
        #pragma unroll
        for (int j = 0; j < 4; ++j) {
            int c = (8 + j) * 16 + lm;
            AFrag b0, b1;
            const _Float16* wp = sW + c * HP;
            b0.p.lo = *(const v8h*)(wp + kh * 16);
            b0.p.hi = *(const v8h*)(wp + kh * 16 + 8);
            b1.p.lo = *(const v8h*)(wp + 32 + kh * 16);
            b1.p.hi = *(const v8h*)(wp + 32 + kh * 16 + 8);
            v8f acc;
            float bh = sBhh[c];
            #pragma unroll
            for (int v = 0; v < 8; ++v) acc[v] = bh;     // gh_n = h@W^T + b_hh
            acc = __builtin_amdgcn_wmma_f32_16x16x32_f16(false, a0.v, false, b0.v, (short)0, acc, false, false);
            acc = __builtin_amdgcn_wmma_f32_16x16x32_f16(false, a1.v, false, b1.v, (short)0, acc, false, false);
            float w0 = sWih0[c], w1 = sWih1[c], bi = sBih[c];
            #pragma unroll
            for (int v = 0; v < 8; ++v) {
                float gi = fmaf(xr[v], w0, fmaf(s2r[v], w1, bi));       // i_n
                float nv = fast_tanh(fmaf(rF[j][v], acc[v], gi));       // tanh(i_n + r*gh_n)
                float z  = zF[j][v];
                float hn = fmaf(z, hD[j][v] - nv, nv);                  // (1-z)*n + z*h
                hD[j][v] = hn;
                int m = v + kh * 8;
                hbuf[m * HP + j * 16 + lm] = (_Float16)hn;              // for next A fragment
            }
        }
    }

    // ---- output projection: out[b] = h . fc_w + fc_b ----
    if (lane < 16) {
        float acc = fc_b[0];
        const _Float16* hp = hbuf + lm * HP;
        #pragma unroll 10
        for (int k = 0; k < 50; ++k)
            acc = fmaf((float)hp[k], sFc[k], acc);
        out[rowbase + lm] = acc;
    }
}

extern "C" void kernel_launch(void* const* d_in, const int* in_sizes, int n_in,
                              void* d_out, int out_size, void* d_ws, size_t ws_size,
                              hipStream_t stream) {
    const float* x     = (const float*)d_in[0];
    const float* omega = (const float*)d_in[1];
    const float* alpha = (const float*)d_in[2];
    const float* beta  = (const float*)d_in[3];
    const float* W_ih  = (const float*)d_in[4];
    const float* W_hh  = (const float*)d_in[5];
    const float* b_ih  = (const float*)d_in[6];
    const float* b_hh  = (const float*)d_in[7];
    const float* fc_w  = (const float*)d_in[8];
    const float* fc_b  = (const float*)d_in[9];
    float* out = (float*)d_out;

    const int B = in_sizes[0] / TSTEPS;          // 2048
    dim3 grid(B / (NW * RPW));                   // 32 blocks
    dim3 block(NW * 32);                         // 128 threads, 4 waves
    hipLaunchKernelGGL(garch_gru_kernel, grid, block, 0, stream,
                       x, omega, alpha, beta, W_ih, W_hh, b_ih, b_hh, fc_w, fc_b, out);
}
